// VectorQuantizer_76338748719692
// MI455X (gfx1250) — compile-verified, matrix operand order checked by reading, not run on hardware
//
#include <hip/hip_runtime.h>

// Problem constants (from reference): B,H,W,D = 16,32,32,256 ; K = 8192
#define N_TOTAL 16384          // B*H*W
#define DDIM    256
#define KCODE   8192
#define TILE_M  64             // rows of z per block
#define CHUNK_N 64             // codebook entries staged in LDS per iteration
#define NCHUNK  (KCODE / CHUNK_N)
#define NSUB    (CHUNK_N / 16) // 4 column sub-tiles per chunk
#define KSTEPS  (DDIM / 32)    // 8 WMMA k-steps (bf16 16x16x32)
#define CHUNK_BYTES (CHUNK_N * DDIM * 2)          // 32 KB
#define CHUNK_B128  (CHUNK_BYTES / 16)            // 2048 x b128
#define LOADS_PER_THREAD (CHUNK_B128 / 128)       // 16

typedef __bf16 bf16_t;
typedef bf16_t v16bf __attribute__((ext_vector_type(16)));
typedef float  v8f   __attribute__((ext_vector_type(8)));

#if defined(__has_builtin)
#if __has_builtin(__builtin_amdgcn_global_load_async_to_lds_b128) && \
    __has_builtin(__builtin_amdgcn_s_wait_asynccnt)
#define HAVE_ASYNC_LDS 1
#endif
#endif

#ifdef HAVE_ASYNC_LDS
// Builtin parameter type per hipcc diagnostic: int __vector_size__(16) pointers.
typedef int v4i __attribute__((vector_size(16)));
typedef __attribute__((address_space(1))) v4i* gbl_v4i_p;   // global source
typedef __attribute__((address_space(3))) v4i* lds_v4i_p;   // LDS destination
#endif

__device__ __forceinline__ unsigned short f32_to_bf16(float f) {
    unsigned u = __float_as_uint(f);
    u += 0x7FFFu + ((u >> 16) & 1u);     // round-to-nearest-even
    return (unsigned short)(u >> 16);
}

// ---------------------------------------------------------------------------
// Kernel 1: convert z and codebook to bf16 in workspace; zero loss scalars.
// ---------------------------------------------------------------------------
__global__ void vq_convert_kernel(const float* __restrict__ z,
                                  const float* __restrict__ cb,
                                  unsigned short* __restrict__ zb,
                                  unsigned short* __restrict__ cbb,
                                  float* __restrict__ scalars) {
    const int gid = blockIdx.x * blockDim.x + threadIdx.x;
    const int NZ4 = N_TOTAL * DDIM / 4;
    const int NC4 = KCODE * DDIM / 4;
    if (gid == 0) { scalars[0] = 0.0f; scalars[1] = 0.0f; }
    if (gid < NZ4) {
        float4 v = ((const float4*)z)[gid];
        ushort4 o;
        o.x = f32_to_bf16(v.x); o.y = f32_to_bf16(v.y);
        o.z = f32_to_bf16(v.z); o.w = f32_to_bf16(v.w);
        ((ushort4*)zb)[gid] = o;
    } else if (gid < NZ4 + NC4) {
        const int i = gid - NZ4;
        float4 v = ((const float4*)cb)[i];
        ushort4 o;
        o.x = f32_to_bf16(v.x); o.y = f32_to_bf16(v.y);
        o.z = f32_to_bf16(v.z); o.w = f32_to_bf16(v.w);
        ((ushort4*)cbb)[i] = o;
    }
}

// ---------------------------------------------------------------------------
// Kernel 2: e2[k] = sum_d codebook[k][d]^2  (f32, one wave32 per row)
// ---------------------------------------------------------------------------
__global__ void vq_e2_kernel(const float* __restrict__ cb,
                             float* __restrict__ e2) {
    const int lane = threadIdx.x & 31;
    const int wave = threadIdx.x >> 5;
    const int row  = blockIdx.x * (blockDim.x >> 5) + wave;
    const float* r = cb + row * DDIM;
    float s = 0.0f;
    #pragma unroll
    for (int i = 0; i < DDIM; i += 32) { float v = r[i + lane]; s += v * v; }
    #pragma unroll
    for (int off = 16; off; off >>= 1) s += __shfl_xor(s, off);
    if (lane == 0) e2[row] = s;
}

// ---------------------------------------------------------------------------
// Kernel 3: WMMA distance GEMM + running argmin.
//  - double-buffered LDS, async global->LDS staging (ASYNCcnt) overlapping
//    chunk i+1 copies with chunk i compute
//  - sub-tiles processed in pairs: two independent WMMA accumulator chains
//    for ILP; epilogue VALU co-executes with the XDL pipe
//  - e2 values for the whole chunk prefetched before the WMMA chains
//   score(n,k) = ||e_k||^2 - 2 * <z_n, e_k>
// ---------------------------------------------------------------------------
__global__ __launch_bounds__(128) void vq_argmin_kernel(
    const unsigned short* __restrict__ zb,
    const unsigned short* __restrict__ cbb,
    const float* __restrict__ e2g,
    int*   __restrict__ idx_ws,
    float* __restrict__ idx_out)
{
    __shared__ uint4 BshRaw[2 * CHUNK_B128];   // 2 x 32 KB double buffer

    const int tid     = threadIdx.x;
    const int lane    = tid & 31;
    const int wave    = tid >> 5;
    const int lmod    = lane & 15;     // column within 16-wide tile / A-row
    const int lhalf   = lane >> 4;     // k-halfgroup (16-bit A/B layout)
    const int rowBase = blockIdx.x * TILE_M + wave * 16;

    // Preload A fragments (16 rows x 256 K bf16) into registers for all of K.
    v16bf a_frag[KSTEPS];
    {
        const unsigned short* zrow =
            zb + (size_t)(rowBase + lmod) * DDIM + lhalf * 16;
        #pragma unroll
        for (int ks = 0; ks < KSTEPS; ++ks)
            a_frag[ks] = *(const v16bf*)(zrow + ks * 32);
    }

    float bestv[8];
    int   besti[8];
    #pragma unroll
    for (int r = 0; r < 8; ++r) { bestv[r] = 3.0e38f; besti[r] = 0; }

#ifdef HAVE_ASYNC_LDS
    // ---- prologue: async-stage chunk 0 into buffer 0 ----
    {
        const char* g = (const char*)cbb;
        char*       l = (char*)BshRaw;
        #pragma unroll
        for (int j = 0; j < LOADS_PER_THREAD; ++j) {
            const int off = (j * 128 + tid) * 16;
            __builtin_amdgcn_global_load_async_to_lds_b128(
                (gbl_v4i_p)(g + off), (lds_v4i_p)(l + off), 0, 0);
        }
    }
#endif

    for (int ci = 0; ci < NCHUNK; ++ci) {
        const int  c0  = ci * CHUNK_N;
        const int  p   = ci & 1;
        bf16_t*    Bsh = (bf16_t*)(BshRaw + p * CHUNK_B128);

        // Prefetch e2 for this chunk's 4 sub-tile columns; latency hides
        // under the staging wait + first WMMA chain.
        float e2v[NSUB];
        #pragma unroll
        for (int n = 0; n < NSUB; ++n)
            e2v[n] = e2g[c0 + n * 16 + lmod];

#ifdef HAVE_ASYNC_LDS
        __builtin_amdgcn_s_wait_asynccnt(0);   // my chunk-ci loads landed
        __syncthreads();                       // everyone's landed; prev buffer free
        if (ci + 1 < NCHUNK) {                 // stage chunk ci+1 into other buffer
            const char* g = (const char*)(cbb + (size_t)(c0 + CHUNK_N) * DDIM);
            char*       l = (char*)(BshRaw + (1 - p) * CHUNK_B128);
            #pragma unroll
            for (int j = 0; j < LOADS_PER_THREAD; ++j) {
                const int off = (j * 128 + tid) * 16;
                __builtin_amdgcn_global_load_async_to_lds_b128(
                    (gbl_v4i_p)(g + off), (lds_v4i_p)(l + off), 0, 0);
            }
        }
#else
        // Fallback: synchronous copy of chunk ci+1 into the other buffer after
        // the barrier; compute still overlaps the loads via the scheduler.
        __syncthreads();                       // chunk-ci stores visible
        if (ci == 0) {                         // prologue copy (chunk 0, buf 0)
            const uint4* src = (const uint4*)cbb;
            #pragma unroll
            for (int j = 0; j < LOADS_PER_THREAD; ++j)
                BshRaw[j * 128 + tid] = src[j * 128 + tid];
            __syncthreads();
        }
        if (ci + 1 < NCHUNK) {
            const uint4* src = (const uint4*)(cbb + (size_t)(c0 + CHUNK_N) * DDIM);
            uint4* dst = BshRaw + (1 - p) * CHUNK_B128;
            #pragma unroll
            for (int j = 0; j < LOADS_PER_THREAD; ++j)
                dst[j * 128 + tid] = src[j * 128 + tid];
        }
#endif

        // Two sub-tiles at a time: independent accumulator chains.
        #pragma unroll
        for (int n = 0; n < NSUB; n += 2) {
            v8f acc0 = {};
            v8f acc1 = {};
            const bf16_t* b0 = Bsh + ((n + 0) * 16 + lmod) * DDIM + lhalf * 16;
            const bf16_t* b1 = Bsh + ((n + 1) * 16 + lmod) * DDIM + lhalf * 16;
            #pragma unroll
            for (int ks = 0; ks < KSTEPS; ++ks) {
                v16bf f0 = *(const v16bf*)(b0 + ks * 32);
                v16bf f1 = *(const v16bf*)(b1 + ks * 32);
                acc0 = __builtin_amdgcn_wmma_f32_16x16x32_bf16(
                        false, a_frag[ks], false, f0, (short)0, acc0, false, false);
                acc1 = __builtin_amdgcn_wmma_f32_16x16x32_bf16(
                        false, a_frag[ks], false, f1, (short)0, acc1, false, false);
            }
            // C layout: VGPR r -> M = r (lanes 0-15) / r+8 (lanes 16-31); N = lane%16
            const int col0 = c0 + (n + 0) * 16 + lmod;
            const int col1 = c0 + (n + 1) * 16 + lmod;
            #pragma unroll
            for (int r = 0; r < 8; ++r) {
                float s0 = __builtin_fmaf(-2.0f, acc0[r], e2v[n + 0]);
                float s1 = __builtin_fmaf(-2.0f, acc1[r], e2v[n + 1]);
                if (s0 < bestv[r]) { bestv[r] = s0; besti[r] = col0; }
                if (s1 < bestv[r]) { bestv[r] = s1; besti[r] = col1; }
            }
        }
    }

    // Reduce over the 16 column-lanes of each half-wave (rows r / r+8).
    #pragma unroll
    for (int r = 0; r < 8; ++r) {
        float bv = bestv[r];
        int   bi = besti[r];
        #pragma unroll
        for (int off = 8; off; off >>= 1) {
            float ov = __shfl_xor(bv, off);
            int   oi = __shfl_xor(bi, off);
            if (ov < bv || (ov == bv && oi < bi)) { bv = ov; bi = oi; }
        }
        if (lmod == 0) {
            const int grow = rowBase + lhalf * 8 + r;
            idx_ws[grow]  = bi;
            idx_out[grow] = (float)bi;   // idx region of d_out (float storage)
        }
    }
}

// ---------------------------------------------------------------------------
// Kernel 4: gather z_q (== forward value of z_st) in f32 + MSE scalars.
// ---------------------------------------------------------------------------
__global__ void vq_gather_loss_kernel(const float* __restrict__ z,
                                      const float* __restrict__ cb,
                                      const int* __restrict__ idx_ws,
                                      float* __restrict__ zst_out,
                                      float* __restrict__ scalars)
{
    __shared__ float wsum[8];
    const int lane = threadIdx.x & 31;
    const int wave = threadIdx.x >> 5;
    const int row  = blockIdx.x * 8 + wave;
    const int code = idx_ws[row];
    const float* zr = z  + (size_t)row  * DDIM;
    const float* er = cb + (size_t)code * DDIM;
    float* zo = zst_out + (size_t)row * DDIM;
    float s = 0.0f;
    #pragma unroll
    for (int i = 0; i < DDIM; i += 32) {
        float e = er[i + lane];
        float d = e - zr[i + lane];
        zo[i + lane] = e;                // z_st forward value == z_q
        s = __builtin_fmaf(d, d, s);
    }
    #pragma unroll
    for (int off = 16; off; off >>= 1) s += __shfl_xor(s, off);
    if (lane == 0) wsum[wave] = s;
    __syncthreads();
    if (threadIdx.x == 0) {
        float t = 0.0f;
        #pragma unroll
        for (int i = 0; i < 8; ++i) t += wsum[i];
        t *= (1.0f / (float)((size_t)N_TOTAL * DDIM));
        atomicAdd(&scalars[0], t);       // commit loss
        atomicAdd(&scalars[1], t);       // codebook loss (same forward value)
    }
}

// ---------------------------------------------------------------------------
extern "C" void kernel_launch(void* const* d_in, const int* in_sizes, int n_in,
                              void* d_out, int out_size, void* d_ws, size_t ws_size,
                              hipStream_t stream) {
    const float* z  = (const float*)d_in[0];   // (16,32,32,256) f32
    const float* cb = (const float*)d_in[1];   // (8192,256) f32

    float* out     = (float*)d_out;
    float* zst     = out;                                    // N*D
    float* idxf    = out + (size_t)N_TOTAL * DDIM;           // N
    float* scalars = idxf + N_TOTAL;                         // commit, codebook_loss

    char* ws = (char*)d_ws;
    unsigned short* zb  = (unsigned short*)ws;                               // 8 MB
    unsigned short* cbb = (unsigned short*)(ws + (size_t)N_TOTAL * DDIM * 2);// 4 MB
    float* e2     = (float*)(ws + (size_t)(N_TOTAL + KCODE) * DDIM * 2);     // 32 KB
    int*   idx_ws = (int*)((char*)e2 + KCODE * sizeof(float));               // 64 KB

    const int totalQuads = (N_TOTAL * DDIM + KCODE * DDIM) / 4;
    vq_convert_kernel<<<(totalQuads + 255) / 256, 256, 0, stream>>>(z, cb, zb, cbb, scalars);
    vq_e2_kernel<<<KCODE / 8, 256, 0, stream>>>(cb, e2);
    vq_argmin_kernel<<<N_TOTAL / TILE_M, 128, 0, stream>>>(zb, cbb, e2, idx_ws, idxf);
    vq_gather_loss_kernel<<<N_TOTAL / 8, 256, 0, stream>>>(z, cb, idx_ws, zst, scalars);
}
